// PSAC_gen_81355270521017
// MI455X (gfx1250) — matrix operand order repaired and checked by reading, not compiled, not run on hardware
//
#include <hip/hip_runtime.h>
#include <hip/hip_bf16.h>
#include <stdint.h>

// Problem constants (B=1)
#define S_LEN 2048
#define L_LEN 3
#define D_DIM 128
#define NKER  16
#define VOCAB 50000
#define K_DIM 256    // 2*D : [Eu | u_attn]
#define PADV  50048  // VOCAB padded to multiple of 64 (zero rows) -> branchless GEMM loop

typedef __attribute__((ext_vector_type(16))) __bf16 v16bf;
typedef __attribute__((ext_vector_type(8)))  float  v8f;

union BF16Frag {
    v16bf v;
    struct { uint4 a, b; } q;   // 2 x 16B = 16 bf16
};

__device__ __forceinline__ uint16_t f2bf_rne(float x) {
    union { float f; uint32_t u; } c; c.f = x;
    uint32_t u = c.u;
    return (uint16_t)((u + 0x7FFFu + ((u >> 16) & 1u)) >> 16);
}

// ------------- Kernel 0: fc2W f32 -> bf16 (RNE), zero-fill pad rows -----------------
__global__ void k_cvt_w(const float* __restrict__ w, uint16_t* __restrict__ wbf) {
    int i = blockIdx.x * blockDim.x + threadIdx.x;
    int stride = gridDim.x * blockDim.x;
    const int nv = VOCAB * K_DIM, np = PADV * K_DIM;
    for (; i < np; i += stride)
        wbf[i] = (i < nv) ? f2bf_rne(w[i]) : (uint16_t)0;
}

// ------------- Kernel 1: Eu = mean over all S*L token embeddings --------------------
__global__ void k_eu(const int* __restrict__ su, const float* __restrict__ emb,
                     float* __restrict__ eu) {
    int t = threadIdx.x;                 // 0..127 == d
    float acc = 0.f;
    for (int idx = 0; idx < S_LEN * L_LEN; ++idx) {
        int tok = su[idx];
        acc += emb[tok * D_DIM + t];
    }
    eu[t] = acc * (1.0f / (float)(S_LEN * L_LEN));
}

// ------------- Kernel 2: per-position encoder (conv + attn + fc1 + user-attn) -------
// One 128-thread block per s. Produces bf16 activation row [Eu | u_attn] (256).
__global__ __launch_bounds__(128) void k_encode(
    const int*   __restrict__ su,   const float* __restrict__ emb,
    const float* __restrict__ convK,const float* __restrict__ convb,
    const float* __restrict__ Wi,   const float* __restrict__ Wj,
    const float* __restrict__ Vv,   const float* __restrict__ fc1W,
    const float* __restrict__ fc1b, const float* __restrict__ eu_g,
    uint16_t* __restrict__ actbf)
{
    __shared__ float ec[L_LEN][D_DIM];
    __shared__ float wi[L_LEN][D_DIM];
    __shared__ float wj[L_LEN][D_DIM];
    __shared__ float red[9][D_DIM];
    __shared__ float alpha[L_LEN][L_LEN];
    __shared__ float att[L_LEN][D_DIM];
    __shared__ float cpart[NKER][8];
    __shared__ float convv[NKER];
    __shared__ float hh[D_DIM];
    __shared__ float eu_s[D_DIM];

    const int s = blockIdx.x;
    const int t = threadIdx.x;           // 0..127

    eu_s[t] = eu_g[t];
    for (int l = 0; l < L_LEN; ++l) {
        int tok = su[s * L_LEN + l];
        ec[l][t] = emb[tok * D_DIM + t];
    }
    __syncthreads();

    // wi[l][e] = Wi[e,:] . ec[l,:] ; wj likewise   (thread t == e)
    {
        const float* wir = Wi + t * D_DIM;
        const float* wjr = Wj + t * D_DIM;
        for (int l = 0; l < L_LEN; ++l) {
            float a0 = 0.f, a1 = 0.f;
            #pragma unroll 4
            for (int d = 0; d < D_DIM; ++d) {
                float e = ec[l][d];
                a0 = fmaf(wir[d], e, a0);
                a1 = fmaf(wjr[d], e, a1);
            }
            wi[l][t] = a0;
            wj[l][t] = a1;
        }
    }
    __syncthreads();

    // scores[i][j] = sum_d Vv[d]*tanh(wi[i][d]+wj[j][d])  (9 reductions of 128)
    {
        float v = Vv[t];
        #pragma unroll
        for (int i = 0; i < 3; ++i)
            #pragma unroll
            for (int j = 0; j < 3; ++j)
                red[i * 3 + j][t] = v * tanhf(wi[i][t] + wj[j][t]);
    }
    __syncthreads();
    if (t < 9) {
        float a = 0.f;
        for (int d = 0; d < D_DIM; ++d) a += red[t][d];
        red[t][0] = a;
    }
    __syncthreads();
    if (t < 3) {  // softmax over j
        float s0 = red[t * 3 + 0][0], s1 = red[t * 3 + 1][0], s2 = red[t * 3 + 2][0];
        float m = fmaxf(s0, fmaxf(s1, s2));
        float e0 = __expf(s0 - m), e1 = __expf(s1 - m), e2 = __expf(s2 - m);
        float inv = 1.f / (e0 + e1 + e2);
        alpha[t][0] = e0 * inv; alpha[t][1] = e1 * inv; alpha[t][2] = e2 * inv;
    }
    __syncthreads();

    // attn[i][d] = sum_j alpha[i][j] * ec[j][d]
    #pragma unroll
    for (int i = 0; i < 3; ++i)
        att[i][t] = alpha[i][0] * ec[0][t] + alpha[i][1] * ec[1][t] + alpha[i][2] * ec[2][t];

    // conv[k] = relu(convK[k,:,:] . ec + convb[k]) : 16 reductions of 384, 8-way split
    {
        int k = t >> 3, c = t & 7;
        const float* kw  = convK + k * (L_LEN * D_DIM);
        const float* ecf = &ec[0][0];
        float a = 0.f;
        int f0 = c * 48;
        for (int f = f0; f < f0 + 48; ++f) a = fmaf(kw[f], ecf[f], a);
        cpart[k][c] = a;
    }
    __syncthreads();
    if (t < NKER) {
        float a = convb[t];
        #pragma unroll
        for (int c = 0; c < 8; ++c) a += cpart[t][c];
        convv[t] = fmaxf(a, 0.f);
    }
    __syncthreads();

    // h[e] = relu(fc1b[e] + fc1W[e,:] . [conv(16) | attn(384)])
    {
        const float* wr = fc1W + t * (NKER + L_LEN * D_DIM);
        float a = fc1b[t];
        #pragma unroll
        for (int k = 0; k < NKER; ++k) a = fmaf(wr[k], convv[k], a);
        const float* af = &att[0][0];
        #pragma unroll 4
        for (int f = 0; f < L_LEN * D_DIM; ++f) a = fmaf(wr[NKER + f], af[f], a);
        hh[t] = fmaxf(a, 0.f);
    }
    __syncthreads();

    // u_attn[i] = sum_j softmax_j(Eu[i]*h[j]) * h[j]   (thread t == i)
    float c0 = eu_s[t];
    float m = -3.4e38f;
    for (int j = 0; j < D_DIM; ++j) m = fmaxf(m, c0 * hh[j]);
    float se = 0.f, sv = 0.f;
    for (int j = 0; j < D_DIM; ++j) {
        float e = __expf(c0 * hh[j] - m);
        se += e;
        sv = fmaf(e, hh[j], sv);
    }
    float u = sv / se;

    // fc2 input row, bf16: [Eu | u_attn]
    actbf[s * K_DIM + t]         = f2bf_rne(eu_s[t]);
    actbf[s * K_DIM + D_DIM + t] = f2bf_rne(u);
}

// ------------- Kernel 3: out = act(2048x256) @ fc2W^T + b, bf16 WMMA ----------------
// Block = 4 waves. Each wave: 2 M16 tiles x 4 N16 tiles (32x64 output, 8 accumulators).
// Weight buffer is zero-padded to PADV rows -> branchless inner loop (EXEC all-ones).
__global__ __launch_bounds__(128) void k_gemm(
    const uint16_t* __restrict__ actbf,   // (2048,256) bf16, K-contiguous
    const uint16_t* __restrict__ wbf,     // (PADV,256) bf16, K-contiguous, zero-padded
    const float*    __restrict__ bias,    // (50000)
    float*          __restrict__ out)     // (2048,50000)
{
    const int lane = threadIdx.x & 31;
    const int wave = threadIdx.x >> 5;            // 0..3
    const int m0   = (blockIdx.y * 4 + wave) * 32;
    const int n0   = blockIdx.x * 64;
    const int lm   = lane & 15;
    const int hk   = lane >> 4;

    v8f acc[2][4] = {};

    // A rows (m-tile 0 and 1), K-contiguous
    const uint16_t* arow0 = actbf + (size_t)(m0 + lm) * K_DIM;
    const uint16_t* arow1 = arow0 + 16 * K_DIM;

    // B rows: lane(n,hk) reads K = {k0+hk*8..+7} and {k0+16+hk*8..+7}
    const uint16_t* brow[4];
    #pragma unroll
    for (int j = 0; j < 4; ++j) {
        const uint16_t* rowbase = wbf + (size_t)(n0 + j * 16 + lm) * K_DIM;
        brow[j] = rowbase + hk * 8;
        // WGP-scope prefetch (locality 3): lanes split by hk cover all four 128B
        // lines of each 512B row across two prefetch instructions.
        __builtin_prefetch(rowbase + hk * 128, 0, 3);
        __builtin_prefetch(rowbase + hk * 128 + 64, 0, 3);
    }

    for (int k0 = 0; k0 < K_DIM; k0 += 32) {
        // A 16x32 frags: lane(m,hk) holds K = {hk*8..+7} and {16+hk*8..+7}
        BF16Frag a0, a1;
        a0.q.a = *(const uint4*)(arow0 + k0 + hk * 8);
        a0.q.b = *(const uint4*)(arow0 + k0 + 16 + hk * 8);
        a1.q.a = *(const uint4*)(arow1 + k0 + hk * 8);
        a1.q.b = *(const uint4*)(arow1 + k0 + 16 + hk * 8);

        #pragma unroll
        for (int j = 0; j < 4; ++j) {
            BF16Frag bf;
            bf.q.a = *(const uint4*)(brow[j] + k0);
            bf.q.b = *(const uint4*)(brow[j] + k0 + 16);
            acc[0][j] = __builtin_amdgcn_wmma_f32_16x16x32_bf16(
                false, a0.v, false, bf.v, (short)0, acc[0][j], false, false);
            acc[1][j] = __builtin_amdgcn_wmma_f32_16x16x32_bf16(
                false, a1.v, false, bf.v, (short)0, acc[1][j], false, false);
        }
    }

    // C layout: element r -> row m + 8*hk + r, col n0 + j*16 + lm
    #pragma unroll
    for (int mt = 0; mt < 2; ++mt) {
        #pragma unroll
        for (int j = 0; j < 4; ++j) {
            int n = n0 + j * 16 + lm;
            if (n >= VOCAB) continue;          // only last block column diverges here
            float b = bias[n];
            float* orow = out + (size_t)(m0 + mt * 16 + 8 * hk) * VOCAB + n;
            #pragma unroll
            for (int r = 0; r < 8; ++r)
                orow[(size_t)r * VOCAB] = acc[mt][j][r] + b;
        }
    }
}

extern "C" void kernel_launch(void* const* d_in, const int* in_sizes, int n_in,
                              void* d_out, int out_size, void* d_ws, size_t ws_size,
                              hipStream_t stream) {
    const int*   su    = (const int*)  d_in[0];
    const float* emb   = (const float*)d_in[1];
    const float* convK = (const float*)d_in[2];
    const float* convb = (const float*)d_in[3];
    const float* Wi    = (const float*)d_in[4];
    const float* Wj    = (const float*)d_in[5];
    const float* Vv    = (const float*)d_in[6];
    const float* fc1W  = (const float*)d_in[7];
    const float* fc1b  = (const float*)d_in[8];
    const float* fc2W  = (const float*)d_in[9];
    const float* fc2b  = (const float*)d_in[10];
    float* out = (float*)d_out;

    // Workspace layout (all 16B aligned):
    //   [0, 25,624,576)          : fc2W bf16, zero-padded (PADV*256*2)
    //   [25,624,576, +512)       : Eu f32      (128*4)
    //   [25,625,088, +1,048,576) : act bf16    (2048*256*2)
    char* ws = (char*)d_ws;
    uint16_t* wbf   = (uint16_t*)ws;
    float*    eu    = (float*)   (ws + 25624576);
    uint16_t* actbf = (uint16_t*)(ws + 25625088);

    k_cvt_w<<<1024, 256, 0, stream>>>(fc2W, wbf);
    k_eu<<<1, 128, 0, stream>>>(su, emb, eu);
    k_encode<<<S_LEN, 128, 0, stream>>>(su, emb, convK, convb, Wi, Wj, Vv,
                                        fc1W, fc1b, eu, actbf);
    dim3 grid(PADV / 64, S_LEN / 128);   // 782 x 16, 4 waves/block
    k_gemm<<<grid, 128, 0, stream>>>(actbf, wbf, fc2b, out);
}
